// SwinEncoder_40475771797520
// MI455X (gfx1250) — matrix operand order repaired and checked by reading, not compiled, not run on hardware
//
#include <hip/hip_runtime.h>
#include <cmath>

// ---------------------------------------------------------------------------
// CDNA5 (gfx1250) Swin encoder. Wave32, WMMA f32<-f16 for all projections.
// Weights pre-transposed to column-major f16 so both WMMA fragments load as
// contiguous 16/32-byte chunks.
// ---------------------------------------------------------------------------

using half_t = _Float16;
typedef __attribute__((ext_vector_type(16))) _Float16 v16h;
typedef __attribute__((ext_vector_type(8)))  _Float16 v8h;
typedef __attribute__((ext_vector_type(8)))  float    v8f;

#define LN_EPS 1e-5f

// ---------------------------------------------------- f32 (K,N) -> f16 (N,K)
__global__ __launch_bounds__(256) void k_cvtT(const float* __restrict__ s,
                                              half_t* __restrict__ d,
                                              int K, int N) {
  int i = blockIdx.x * 256 + threadIdx.x;
  if (i >= K * N) return;
  int n = i % N, k = i / N;               // i indexes source row-major (K,N)
  d[(long long)n * K + k] = (half_t)s[i];
}

// ------------------------------------------------------------- direct conv2d
// Output is BHWC. Input layout selectable (patch-embed input is BCHW).
__global__ __launch_bounds__(256) void k_conv2d(
    const float* __restrict__ x, const float* __restrict__ w,
    const float* __restrict__ bias, float* __restrict__ y,
    int Bn, int Cin, int Hin, int Win, int Cout, int Kh, int Kw, int stride,
    int Hout, int Wout, int in_bchw) {
  long long idx = (long long)blockIdx.x * 256 + threadIdx.x;
  long long total = (long long)Bn * Hout * Wout * Cout;
  if (idx >= total) return;
  int co = idx % Cout; long long t = idx / Cout;
  int ow = t % Wout; t /= Wout;
  int oh = t % Hout; int b = (int)(t / Hout);
  float acc = bias[co];
  for (int kh = 0; kh < Kh; ++kh) {
    int ih = oh * stride + kh;
    for (int kw = 0; kw < Kw; ++kw) {
      int iw = ow * stride + kw;
      const float* wr = w + ((long long)co * Cin) * Kh * Kw + kh * Kw + kw;
      for (int ci = 0; ci < Cin; ++ci) {
        float xv = in_bchw
            ? x[(((long long)b * Cin + ci) * Hin + ih) * Win + iw]
            : x[(((long long)b * Hin + ih) * Win + iw) * Cin + ci];
        acc = fmaf(xv, wr[(long long)ci * Kh * Kw], acc);
      }
    }
  }
  y[idx] = acc;
}

// ----------------------------------------------- LayerNorm (+shift+window)
// One wave per token. windowed=1: write rows in (window, n) order after the
// forward roll by sfwd (faithful -ws//2). windowed=0: plain token order.
__global__ __launch_bounds__(256) void k_ln(
    const float* __restrict__ x, const float* __restrict__ gamma,
    const float* __restrict__ beta, half_t* __restrict__ out,
    int Bn, int H, int W, int C, int ws, int sfwd, int windowed) {
  int t = blockIdx.x * 8 + (threadIdx.x >> 5);
  if (t >= Bn * H * W) return;
  int lane = threadIdx.x & 31;
  int b = t / (H * W);
  int rem = t - b * H * W;
  int h = rem / W, w = rem - (rem / W) * W;
  const float* xr = x + (long long)t * C;
  float vals[8];
  int cnt = 0;
  float s = 0.f;
  for (int c = lane; c < C; c += 32) { float v = xr[c]; vals[cnt++] = v; s += v; }
  for (int o = 16; o; o >>= 1) s += __shfl_xor(s, o, 32);
  float mu = s / (float)C;
  float ss = 0.f; cnt = 0;
  for (int c = lane; c < C; c += 32) { float d = vals[cnt++] - mu; ss += d * d; }
  for (int o = 16; o; o >>= 1) ss += __shfl_xor(ss, o, 32);
  float inv = rsqrtf(ss / (float)C + LN_EPS);
  long long row;
  if (windowed) {
    int hh = h - sfwd; if (hh < 0) hh += H;
    int ww = w - sfwd; if (ww < 0) ww += W;
    int win = (b * (H / ws) + hh / ws) * (W / ws) + ww / ws;
    row = (long long)win * ws * ws + (hh % ws) * ws + (ww % ws);
  } else {
    row = t;
  }
  half_t* orow = out + row * C;
  cnt = 0;
  for (int c = lane; c < C; c += 32)
    orow[c] = (half_t)((vals[cnt++] - mu) * inv * gamma[c] + beta[c]);
}

// ------------------------------------------------------------- WMMA GEMM
// D[M,N] = act(A[M,K]f16 * Bt[N,K]f16^T + bias) (+resid). One wave per 16x16
// C tile, 8 waves per block along M, K stepped 32 via v_wmma_f32_16x16x32_f16.
// Bt is COLUMN-major (N,K) so both fragments are contiguous vector loads.
// Fragment layouts per CDNA5 ISA 7.12.2 (wave32).
__global__ __launch_bounds__(256) void k_gemm_wmma(
    const half_t* __restrict__ A, const half_t* __restrict__ Bt,
    const float* __restrict__ bias, float* __restrict__ out32,
    half_t* __restrict__ out16, const float* __restrict__ resid,
    int M, int N, int K, int act) {
  const int lane = threadIdx.x & 31;
  const int wave = threadIdx.x >> 5;
  const int n0 = blockIdx.x * 16;
  const int m0 = (blockIdx.y * 8 + wave) * 16;
  if (m0 >= M) return;
  const int hid = lane >> 4;   // half-wave id
  const int l16 = lane & 15;
  v8f acc = {};
  // A: lane row m0+l16; halves K = hid*8..+7 and hid*8+16..+23 (contiguous).
  const half_t* Ar = A + (long long)(m0 + l16) * K + hid * 8;
  // B: lane col n0+l16; halves K = hid*8..+7 and hid*8+16..+23 (contiguous in Bt).
  const half_t* Br = Bt + (long long)(n0 + l16) * K + hid * 8;
#pragma unroll 2
  for (int kb = 0; kb < K; kb += 32) {
    v8h alo = *(const v8h*)(Ar + kb);
    v8h ahi = *(const v8h*)(Ar + kb + 16);
    v8h blo = *(const v8h*)(Br + kb);
    v8h bhi = *(const v8h*)(Br + kb + 16);
    v16h a, b;
#pragma unroll
    for (int i = 0; i < 8; ++i) { a[i] = alo[i]; a[i + 8] = ahi[i]; b[i] = blo[i]; b[i + 8] = bhi[i]; }
    acc = __builtin_amdgcn_wmma_f32_16x16x32_f16(
        false, a, false, b, (short)0, acc, false, false);
  }
#pragma unroll
  for (int r = 0; r < 8; ++r) {            // C/D: VGPR r -> M = hid*8 + r, N = l16
    int m = m0 + hid * 8 + r;
    int n = n0 + l16;
    float v = acc[r];
    if (bias) v += bias[n];
    if (act == 1) v = 0.5f * v * (1.0f + erff(v * 0.70710678118654752f));
    long long o = (long long)m * N + n;
    if (resid) v += resid[o];
    if (out32) out32[o] = v;
    if (out16) out16[o] = (half_t)v;
  }
}

// ----------------------------------------------------- window attention core
// d = C/heads = 32 for every stage of this net. One thread per (win, head, n).
template <int WS>
__global__ __launch_bounds__(128) void k_attn(
    const half_t* __restrict__ qkv, const float* __restrict__ relb,
    half_t* __restrict__ out, int nWin, int C, int heads) {
  constexpr int N = WS * WS;
  int idx = blockIdx.x * 128 + threadIdx.x;
  if (idx >= nWin * heads * N) return;
  int n = idx % N; int t = idx / N;
  int h = t % heads; int win = t / heads;
  const int d = 32;
  const int C3 = 3 * C;
  long long base = (long long)win * N * C3;
  const half_t* qrow = qkv + base + (long long)n * C3 + h * d;
  float q[32];
  const float scale = 0.17677669529663689f;  // 32^-0.5
#pragma unroll
  for (int j = 0; j < 32; ++j) q[j] = (float)qrow[j] * scale;
  float sc[N];
  float mx = -1e30f;
  int i0 = n / WS, j0 = n % WS;
#pragma unroll
  for (int m = 0; m < N; ++m) {
    const half_t* krow = qkv + base + (long long)m * C3 + C + h * d;
    float s = 0.f;
#pragma unroll
    for (int j = 0; j < 32; ++j) s += q[j] * (float)krow[j];
    int i1 = m / WS, j1 = m % WS;
    int ridx = (i0 - i1 + WS - 1) * (2 * WS - 1) + (j0 - j1 + WS - 1);
    s += relb[ridx * heads + h];
    sc[m] = s;
    mx = fmaxf(mx, s);
  }
  float sum = 0.f;
#pragma unroll
  for (int m = 0; m < N; ++m) { float e = __expf(sc[m] - mx); sc[m] = e; sum += e; }
  float rs = 1.f / sum;
  half_t* orow = out + ((long long)win * N + n) * C + h * d;
#pragma unroll
  for (int j = 0; j < 32; ++j) {
    float a = 0.f;
#pragma unroll
    for (int m = 0; m < N; ++m)
      a += sc[m] * (float)qkv[base + (long long)m * C3 + 2 * C + h * d + j];
    orow[j] = (half_t)(a * rs);
  }
}

// ------------------------------------------- un-window (+reverse roll by sb)
__global__ __launch_bounds__(256) void k_unwindow(
    const float* __restrict__ src, float* __restrict__ dst,
    int Bn, int H, int W, int C, int ws, int sb) {
  long long idx = (long long)blockIdx.x * 256 + threadIdx.x;
  long long total = (long long)Bn * H * W * C;
  if (idx >= total) return;
  int c = idx % C; long long t = idx / C;
  int w = t % W; t /= W;
  int h = t % H; int b = (int)(t / H);
  int hh = h - sb; if (hh < 0) hh += H;
  int ww = w - sb; if (ww < 0) ww += W;
  long long row = ((long long)(b * (H / ws) + hh / ws) * (W / ws) + ww / ws) *
                      (ws * ws) + (hh % ws) * ws + (ww % ws);
  dst[idx] = src[row * C + c];
}

// ------------------------------------------------------------ BHWC -> BCHW
__global__ __launch_bounds__(256) void k_to_bchw(
    const float* __restrict__ src, float* __restrict__ dst,
    int Bn, int H, int W, int C) {
  long long idx = (long long)blockIdx.x * 256 + threadIdx.x;
  long long total = (long long)Bn * C * H * W;
  if (idx >= total) return;
  int w = idx % W; long long t = idx / W;
  int h = t % H; t /= H;
  int c = t % C; int b = (int)(t / C);
  dst[idx] = src[(((long long)b * H + h) * W + w) * C + c];
}

// ---------------------------------------------------------------------------
// Host orchestration
// ---------------------------------------------------------------------------
namespace {
struct BW {
  const float *ln1g, *ln1b, *qkvw, *relb, *projw, *projb;
  const float *ln2g, *ln2b, *fc1w, *fc1b, *fc2w, *fc2b;
  half_t *qkv16, *proj16, *fc116, *fc216;   // column-major (N,K) f16
  int C;
};
}  // namespace

extern "C" void kernel_launch(void* const* d_in, const int* in_sizes, int n_in,
                              void* d_out, int out_size, void* d_ws,
                              size_t ws_size, hipStream_t stream) {
  (void)in_sizes; (void)n_in; (void)out_size;
  const int Bn = 16;
  const long long MAX_TC = 16LL * 96 * 96 * 64;  // largest tokens*C (stage 1)

  // ---- workspace carve-up ----
  size_t off = 0;
  auto carve = [&](size_t bytes) {
    size_t o = off;
    off += (bytes + 255) & ~(size_t)255;
    return o;
  };
  char* base = (char*)d_ws;
  size_t oF0   = carve(MAX_TC * 4);
  size_t oF1   = carve(MAX_TC * 4);
  size_t oF2   = carve(MAX_TC * 4);
  size_t oLN   = carve(MAX_TC * 2);      // f16 LN rows
  size_t oQKV  = carve(MAX_TC * 3 * 2);  // f16 qkv rows
  size_t oAT   = carve(MAX_TC * 2);      // f16 attn out rows
  size_t oH    = carve(MAX_TC * 4 * 2);  // f16 fc1 rows
  size_t oW16  = carve(2100000 * 2);     // f16 weight pool (transposed)
  if (off > ws_size) return;             // workspace too small: no-op

  float*  F[3]  = {(float*)(base + oF0), (float*)(base + oF1), (float*)(base + oF2)};
  half_t* LN16  = (half_t*)(base + oLN);
  half_t* QKV16 = (half_t*)(base + oQKV);
  half_t* AT16  = (half_t*)(base + oAT);
  half_t* H16   = (half_t*)(base + oH);
  half_t* WPOOL = (half_t*)(base + oW16);

  // ---- parse inputs (setup_inputs() insertion order) ----
  int p = 0;
  const float* x_in = (const float*)d_in[p++];            // (16,20,384,384)
  const float* pe_w = (const float*)d_in[p++];
  const float* pe_b = (const float*)d_in[p++];
  BW blk[6];
  auto readBlock = [&](BW& b, int C) {
    b.ln1g = (const float*)d_in[p++]; b.ln1b = (const float*)d_in[p++];
    b.qkvw = (const float*)d_in[p++]; b.relb = (const float*)d_in[p++];
    b.projw = (const float*)d_in[p++]; b.projb = (const float*)d_in[p++];
    b.ln2g = (const float*)d_in[p++]; b.ln2b = (const float*)d_in[p++];
    b.fc1w = (const float*)d_in[p++]; b.fc1b = (const float*)d_in[p++];
    b.fc2w = (const float*)d_in[p++]; b.fc2b = (const float*)d_in[p++];
    b.C = C;
  };
  readBlock(blk[0], 64);  readBlock(blk[1], 64);
  const float* d1w = (const float*)d_in[p++];
  const float* d1b = (const float*)d_in[p++];
  readBlock(blk[2], 128); readBlock(blk[3], 128);
  const float* d2w = (const float*)d_in[p++];
  const float* d2b = (const float*)d_in[p++];
  readBlock(blk[4], 256); readBlock(blk[5], 256);

  // ---- convert all GEMM weights to transposed f16 once ----
  size_t woff = 0;
  auto cvtT = [&](const float* src, int K, int N) -> half_t* {
    half_t* dst = WPOOL + woff;
    size_t n = (size_t)K * N;
    woff += n;
    k_cvtT<<<dim3((unsigned)((n + 255) / 256)), dim3(256), 0, stream>>>(src, dst, K, N);
    return dst;
  };
  for (int i = 0; i < 6; ++i) {
    int C = blk[i].C;
    blk[i].qkv16  = cvtT(blk[i].qkvw, C, 3 * C);
    blk[i].proj16 = cvtT(blk[i].projw, C, C);
    blk[i].fc116  = cvtT(blk[i].fc1w, C, 4 * C);
    blk[i].fc216  = cvtT(blk[i].fc2w, 4 * C, C);
  }

  auto gemm = [&](const half_t* A, const half_t* Bt, const float* bias,
                  float* o32, half_t* o16, const float* resid,
                  int M, int N, int K, int act) {
    dim3 g((unsigned)((N + 15) / 16), (unsigned)((M / 16 + 7) / 8));
    k_gemm_wmma<<<g, dim3(256), 0, stream>>>(A, Bt, bias, o32, o16, resid,
                                             M, N, K, act);
  };
  auto attn = [&](int ws, const float* relb, int nWin, int C, int heads) {
    int N = ws * ws;
    int tot = nWin * heads * N;
    dim3 g((unsigned)((tot + 127) / 128));
    if (ws == 6)      k_attn<6><<<g, dim3(128), 0, stream>>>(QKV16, relb, AT16, nWin, C, heads);
    else if (ws == 3) k_attn<3><<<g, dim3(128), 0, stream>>>(QKV16, relb, AT16, nWin, C, heads);
    else              k_attn<2><<<g, dim3(128), 0, stream>>>(QKV16, relb, AT16, nWin, C, heads);
  };

  // One Swin block: xin (BHWC f32) -> xin, using s1/s2 as f32 scratch.
  auto run_block = [&](float* xin, float* s1, float* s2, const BW& bw,
                       int H, int W, int heads, int ws, bool shift) {
    int C = bw.C;
    int tokens = Bn * H * W;
    int nWin = Bn * (H / ws) * (W / ws);
    int sfwd = shift ? (ws + 1) / 2 : 0;  // roll(-ws//2) faithful
    int sb   = shift ? ws / 2 : 0;        // roll back by ws//2
    unsigned lnG = (unsigned)((tokens + 7) / 8);
    unsigned eG  = (unsigned)(((long long)tokens * C + 255) / 256);
    // LN1 + shift + window partition -> f16 rows
    k_ln<<<lnG, dim3(256), 0, stream>>>(xin, bw.ln1g, bw.ln1b, LN16,
                                        Bn, H, W, C, ws, sfwd, 1);
    // qkv (no bias)
    gemm(LN16, bw.qkv16, nullptr, nullptr, QKV16, nullptr, tokens, 3 * C, C, 0);
    // softmax(QK^T * scale + rel_bias) V
    attn(ws, bw.relb, nWin, C, heads);
    // proj (+bias) -> window-ordered f32
    gemm(AT16, bw.proj16, bw.projb, s1, nullptr, nullptr, tokens, C, C, 0);
    // un-window + reverse roll -> new x (no attention residual, per reference)
    k_unwindow<<<eG, dim3(256), 0, stream>>>(s1, s2, Bn, H, W, C, ws, sb);
    // LN2 (plain token order)
    k_ln<<<lnG, dim3(256), 0, stream>>>(s2, bw.ln2g, bw.ln2b, LN16,
                                        Bn, H, W, C, 1, 0, 0);
    // fc1 + bias + exact GELU -> f16
    gemm(LN16, bw.fc116, bw.fc1b, nullptr, H16, nullptr, tokens, 4 * C, C, 1);
    // fc2 + bias + residual(x) -> block output
    gemm(H16, bw.fc216, bw.fc2b, xin, nullptr, s2, tokens, C, 4 * C, 0);
  };

  float* out = (float*)d_out;

  // ---- patch embed: (16,20,384,384) BCHW -> F0 (16,96,96,64) BHWC ----
  {
    long long tot = (long long)Bn * 96 * 96 * 64;
    k_conv2d<<<dim3((unsigned)((tot + 255) / 256)), dim3(256), 0, stream>>>(
        x_in, pe_w, pe_b, F[0], Bn, 20, 384, 384, 64, 4, 4, 4, 96, 96, 1);
  }
  // ---- stage 1: dim 64, heads 2, ws 6 ----
  run_block(F[0], F[1], F[2], blk[0], 96, 96, 2, 6, false);
  run_block(F[0], F[1], F[2], blk[1], 96, 96, 2, 6, true);
  {
    long long tot = (long long)Bn * 64 * 96 * 96;
    k_to_bchw<<<dim3((unsigned)((tot + 255) / 256)), dim3(256), 0, stream>>>(
        F[0], out, Bn, 96, 96, 64);
  }
  // ---- downsample 1: 2x2 s2 -> F1 (16,48,48,128) BHWC ----
  {
    long long tot = (long long)Bn * 48 * 48 * 128;
    k_conv2d<<<dim3((unsigned)((tot + 255) / 256)), dim3(256), 0, stream>>>(
        F[0], d1w, d1b, F[1], Bn, 64, 96, 96, 128, 2, 2, 2, 48, 48, 0);
  }
  // ---- stage 2: dim 128, heads 4, ws 3 ----
  run_block(F[1], F[0], F[2], blk[2], 48, 48, 4, 3, false);
  run_block(F[1], F[0], F[2], blk[3], 48, 48, 4, 3, true);
  {
    long long tot = (long long)Bn * 128 * 48 * 48;
    k_to_bchw<<<dim3((unsigned)((tot + 255) / 256)), dim3(256), 0, stream>>>(
        F[1], out + 9437184LL, Bn, 48, 48, 128);
  }
  // ---- downsample 2: 3x3 s3 -> F0 (16,16,16,256) BHWC (16%2==0: no pad) ----
  {
    long long tot = (long long)Bn * 16 * 16 * 256;
    k_conv2d<<<dim3((unsigned)((tot + 255) / 256)), dim3(256), 0, stream>>>(
        F[1], d2w, d2b, F[0], Bn, 128, 48, 48, 256, 3, 3, 3, 16, 16, 0);
  }
  // ---- stage 3: dim 256, heads 8, ws 2 ----
  run_block(F[0], F[1], F[2], blk[4], 16, 16, 8, 2, false);
  run_block(F[0], F[1], F[2], blk[5], 16, 16, 8, 2, true);
  {
    long long tot = (long long)Bn * 256 * 16 * 16;
    k_to_bchw<<<dim3((unsigned)((tot + 255) / 256)), dim3(256), 0, stream>>>(
        F[0], out + 9437184LL + 4718592LL, Bn, 16, 16, 256);
  }
}
